// MambaWithLuenbergerObserver_46145128628831
// MI455X (gfx1250) — matrix operand order repaired and checked by reading, not compiled, not run on hardware
//
#include <hip/hip_runtime.h>
#include <hip/hip_bf16.h>
#include <math.h>

#define BATCH 2
#define LSEQ  2048
#define DMOD  1024
#define DIN   2048      // DI = 2*DM
#define NST   16
#define NAUG  32        // N2
#define KCONV 4
#define DTR   64
#define ALPHA 0.1f

typedef __bf16 bf16_t;
typedef __attribute__((ext_vector_type(16))) __bf16 v16bf;
typedef __attribute__((ext_vector_type(8)))  __bf16 v8bf;
typedef __attribute__((ext_vector_type(8)))  float  v8f;

__device__ __forceinline__ float softplus_f(float x) {
  return (x > 20.f) ? x : log1pf(__expf(x));
}
__device__ __forceinline__ float silu_f(float x) {
  return x / (1.f + __expf(-x));
}
__device__ __forceinline__ void split_bf16(float x, bf16_t& h, bf16_t& l) {
  h = (bf16_t)x;
  l = (bf16_t)(x - (float)h);
}

// ---------------------------------------------------------------------------
// One-time fp32 -> bf16 hi/lo split (for GEMM operands)
// ---------------------------------------------------------------------------
__global__ void cvt_kernel(const float* __restrict__ src,
                           bf16_t* __restrict__ hi, bf16_t* __restrict__ lo)
{
  size_t idx = (size_t)blockIdx.x * blockDim.x + threadIdx.x;
  bf16_t h, l;
  split_bf16(src[idx], h, l);
  hi[idx] = h; lo[idx] = l;
}

// ---------------------------------------------------------------------------
// C[M,N] = A[M,K] * B[N,K]^T with bf16x3 split (near-fp32 accuracy):
//   acc += Alo*Bhi + Ahi*Blo + Ahi*Bhi   (fp32 WMMA accumulate)
// Operands are pre-split bf16 hi/lo, row-major [row][k] -> fragments load
// straight from global (contiguous-K per lane), no LDS staging needed.
// 8 waves/block, wave tile = 32(M) x 64(N): A frags reused x4, B frags x2.
// Optionally also emits a bf16 hi/lo copy of C (for GEMM2 -> GEMM3 chain).
// ---------------------------------------------------------------------------
__global__ __launch_bounds__(256) void gemm_bf16x3_kernel(
    const bf16_t* __restrict__ Ah, const bf16_t* __restrict__ Al, int lda,
    const bf16_t* __restrict__ Bh, const bf16_t* __restrict__ Bl, int ldb,
    float* __restrict__ C, int ldc,
    bf16_t* __restrict__ Ch, bf16_t* __restrict__ Cl,
    int N, int K)
{
  const int lane = threadIdx.x & 31;
  const int wid  = threadIdx.x >> 5;
  const int tilesN = N >> 6;
  const int tile = blockIdx.x * 8 + wid;
  const int tM = tile / tilesN;            // 32-row tile index
  const int tN = tile % tilesN;            // 64-col tile index

  const int fm = lane & 15;                // row within 16-wide subtile
  const int fh = lane >> 4;                // k-half / m-half selector

  v8f acc[2][4];
  for (int mt = 0; mt < 2; ++mt)
    for (int nt = 0; nt < 4; ++nt)
      for (int i = 0; i < 8; ++i) acc[mt][nt][i] = 0.f;

  for (int k0 = 0; k0 < K; k0 += 32) {
    // A fragments (16-bit A 16x32 layout: elems 0..7 -> K=k0+fh*8..,
    // elems 8..15 -> K=k0+16+fh*8..)
    v16bf ah[2], al[2];
    for (int mt = 0; mt < 2; ++mt) {
      const size_t arow = (size_t)(tM * 32 + mt * 16 + fm) * lda + k0 + fh * 8;
      const v8bf h0 = *(const v8bf*)(Ah + arow);
      const v8bf h1 = *(const v8bf*)(Ah + arow + 16);
      const v8bf l0 = *(const v8bf*)(Al + arow);
      const v8bf l1 = *(const v8bf*)(Al + arow + 16);
      for (int i = 0; i < 8; ++i) {
        ah[mt][i] = h0[i]; ah[mt][8 + i] = h1[i];
        al[mt][i] = l0[i]; al[mt][8 + i] = l1[i];
      }
    }
    if (k0 + 32 < K) {
      __builtin_prefetch(Ah + (size_t)(tM * 32 + fm) * lda + k0 + 32, 0, 0);
      __builtin_prefetch(Bh + (size_t)(tN * 64 + (lane << 1)) * ldb + k0 + 32, 0, 0);
    }
    for (int nt = 0; nt < 4; ++nt) {
      // B fragment: lane fm = column n; elems 0..7 -> K=k0+fh*8.., 8..15 -> K=k0+16+fh*8..
      const size_t brow = (size_t)(tN * 64 + nt * 16 + fm) * ldb + k0 + fh * 8;
      const v8bf h0 = *(const v8bf*)(Bh + brow);
      const v8bf h1 = *(const v8bf*)(Bh + brow + 16);
      const v8bf l0 = *(const v8bf*)(Bl + brow);
      const v8bf l1 = *(const v8bf*)(Bl + brow + 16);
      v16bf bh, bl;
      for (int i = 0; i < 8; ++i) {
        bh[i] = h0[i]; bh[8 + i] = h1[i];
        bl[i] = l0[i]; bl[8 + i] = l1[i];
      }
      for (int mt = 0; mt < 2; ++mt) {
        acc[mt][nt] = __builtin_amdgcn_wmma_f32_16x16x32_bf16(
            false, al[mt], false, bh, (short)0, acc[mt][nt], false, false);
        acc[mt][nt] = __builtin_amdgcn_wmma_f32_16x16x32_bf16(
            false, ah[mt], false, bl, (short)0, acc[mt][nt], false, false);
        acc[mt][nt] = __builtin_amdgcn_wmma_f32_16x16x32_bf16(
            false, ah[mt], false, bh, (short)0, acc[mt][nt], false, false);
      }
    }
  }

  // C/D layout: VGPR r, lane l -> M = r + 8*(l>>4), N = l&15
  for (int mt = 0; mt < 2; ++mt) {
    for (int nt = 0; nt < 4; ++nt) {
      for (int r = 0; r < 8; ++r) {
        const size_t m = (size_t)(tM * 32 + mt * 16 + r + 8 * fh);
        const size_t idx = m * ldc + tN * 64 + nt * 16 + fm;
        const float v = acc[mt][nt][r];
        C[idx] = v;
        if (Ch != nullptr) {
          bf16_t h, l;
          split_bf16(v, h, l);
          Ch[idx] = h; Cl[idx] = l;
        }
      }
    }
  }
}

// ---------------------------------------------------------------------------
// gamma = softplus(observer_gain); dmean = mean(D)
// ---------------------------------------------------------------------------
__global__ void prep_kernel(const float* __restrict__ og,
                            const float* __restrict__ Dv,
                            float* __restrict__ gamma,
                            float* __restrict__ dmean)
{
  int t = threadIdx.x;
  if (t < NST) gamma[t] = softplus_f(og[t]);
  if (t == 0) {
    float s = 0.f;
    for (int i = 0; i < DIN; ++i) s += Dv[i];
    dmean[0] = s / (float)DIN;
  }
}

// ---------------------------------------------------------------------------
// A_aug[d,n] = -exp(A_log[d, n&15]) - (n>=16 ? gamma[n-16] : 0)
// ---------------------------------------------------------------------------
__global__ void aaug_kernel(const float* __restrict__ A_log,
                            const float* __restrict__ gamma,
                            float* __restrict__ Aaug)
{
  int idx = blockIdx.x * blockDim.x + threadIdx.x;   // DIN*32
  int n = idx & 31;
  int d = idx >> 5;
  int n16 = n & 15;
  float a = -__expf(A_log[(size_t)d * 32 + n16]);
  Aaug[idx] = (n < NST) ? a : (a - gamma[n16]);
}

// ---------------------------------------------------------------------------
// causal depthwise conv1d (K=4) + bias + SiLU; emits f32 (for scan) and
// pre-split bf16 hi/lo (as GEMM2's A operand).
// ---------------------------------------------------------------------------
__global__ void conv_silu_kernel(const float* __restrict__ xz,      // (B*L, 4096)
                                 const float* __restrict__ conv_w,  // (DIN, 4)
                                 const float* __restrict__ conv_b,
                                 float* __restrict__ xconv,         // (B*L, DIN)
                                 bf16_t* __restrict__ xch,
                                 bf16_t* __restrict__ xcl)
{
  int idx = blockIdx.x * blockDim.x + threadIdx.x;   // B*L*DIN
  int d = idx & (DIN - 1);
  int row = idx >> 11;                               // b*L + l
  int l = row & (LSEQ - 1);
  float acc = conv_b[d];
  #pragma unroll
  for (int k = 0; k < KCONV; ++k) {
    int lp = l + k - (KCONV - 1);
    if (lp >= 0)
      acc += conv_w[d * KCONV + k] *
             xz[(size_t)(row + k - (KCONV - 1)) * 4096 + d];
  }
  float y = silu_f(acc);
  xconv[(size_t)idx] = y;
  bf16_t h, lo;
  split_bf16(y, h, lo);
  xch[(size_t)idx] = h; xcl[(size_t)idx] = lo;
}

// ---------------------------------------------------------------------------
// B_aug / C_aug from x_dbl columns [64:96) and [96:128)
// ---------------------------------------------------------------------------
__global__ void bcaug_kernel(const float* __restrict__ xdbl,  // (B*L, 128)
                             const float* __restrict__ gamma,
                             const float* __restrict__ dmean,
                             float* __restrict__ Baug,        // (B*L, 32)
                             float* __restrict__ Caug)        // (B*L, 32)
{
  int idx = blockIdx.x * blockDim.x + threadIdx.x;   // (B*L)*32
  int n = idx & 31;
  size_t row = (size_t)(idx >> 5);
  int n16 = n & 15;
  float Bo = xdbl[row * 128 + DTR + n16];
  float Co = xdbl[row * 128 + DTR + NAUG + n16];
  Baug[idx] = (n < NST) ? Bo : (Bo + gamma[n16] * dmean[0]);
  Caug[idx] = (n < NST) ? (1.f - ALPHA) * Co : ALPHA * Co;
}

// ---------------------------------------------------------------------------
// delta = softplus(dt_pre + dt_proj_b[d])   (in place)
// ---------------------------------------------------------------------------
__global__ void delta_kernel(float* __restrict__ dpre,
                             const float* __restrict__ dtb)
{
  int idx = blockIdx.x * blockDim.x + threadIdx.x;   // B*L*DIN
  int d = idx & (DIN - 1);
  dpre[(size_t)idx] = softplus_f(dpre[(size_t)idx] + dtb[d]);
}

// ---------------------------------------------------------------------------
// Selective scan: one wave per (b,d) channel; lane = augmented state index n.
// s_n <- s_n*exp(dt*A[d,n]) + dt*u*B[l,n];  y = sum_n s_n*C[l,n] (shfl-xor).
// ---------------------------------------------------------------------------
__global__ __launch_bounds__(256) void scan_kernel(
    const float* __restrict__ delta,   // (B*L, DIN)
    const float* __restrict__ u,       // (B*L, DIN)
    const float* __restrict__ Baug,    // (B*L, 32)
    const float* __restrict__ Caug,    // (B*L, 32)
    const float* __restrict__ Aaug,    // (DIN, 32)
    float* __restrict__ yscan)         // (B*L, DIN)
{
  const int lane = threadIdx.x & 31;
  const int wid  = threadIdx.x >> 5;
  const int gidx = blockIdx.x * 8 + wid;   // (b,d) pair
  const int b = gidx / DIN;
  const int d = gidx % DIN;
  const float Areg = Aaug[(size_t)d * 32 + lane];
  float s = 0.f;
  for (int l = 0; l < LSEQ; ++l) {
    const size_t row = (size_t)b * LSEQ + l;
    float dt = delta[row * DIN + d];
    float ut = u[row * DIN + d];
    float Bv = Baug[row * 32 + lane];
    float Cv = Caug[row * 32 + lane];
    s = s * __expf(dt * Areg) + dt * ut * Bv;
    float p = s * Cv;
    for (int off = 16; off > 0; off >>= 1)
      p += __shfl_xor(p, off, 32);
    if (lane == 0) yscan[row * DIN + d] = p;
  }
}

// ---------------------------------------------------------------------------
// ygated = (y_scan + D[d]*u) * silu(z), emitted directly as bf16 hi/lo
// (it is only consumed as GEMM4's A operand).
// ---------------------------------------------------------------------------
__global__ void gate_kernel(const float* __restrict__ yscan,
                            const float* __restrict__ xconv,
                            const float* __restrict__ xz,
                            const float* __restrict__ Dv,
                            bf16_t* __restrict__ ygh,
                            bf16_t* __restrict__ ygl)
{
  int idx = blockIdx.x * blockDim.x + threadIdx.x;   // B*L*DIN
  int d = idx & (DIN - 1);
  size_t row = (size_t)(idx >> 11);
  float uu = xconv[(size_t)idx];
  float y  = yscan[(size_t)idx] + Dv[d] * uu;
  float z  = xz[row * 4096 + DIN + d];
  float g  = y * silu_f(z);
  bf16_t h, l;
  split_bf16(g, h, l);
  ygh[(size_t)idx] = h; ygl[(size_t)idx] = l;
}

// ---------------------------------------------------------------------------
extern "C" void kernel_launch(void* const* d_in, const int* in_sizes, int n_in,
                              void* d_out, int out_size, void* d_ws, size_t ws_size,
                              hipStream_t stream)
{
  const float* hidden     = (const float*)d_in[0];
  const float* in_proj_w  = (const float*)d_in[1];
  const float* conv_w     = (const float*)d_in[2];
  const float* conv_b     = (const float*)d_in[3];
  const float* x_proj_w   = (const float*)d_in[4];
  const float* dt_proj_w  = (const float*)d_in[5];
  const float* dt_proj_b  = (const float*)d_in[6];
  const float* A_log      = (const float*)d_in[7];
  const float* Dv         = (const float*)d_in[8];
  const float* out_proj_w = (const float*)d_in[9];
  const float* obs_gain   = (const float*)d_in[10];

  const size_t BL = (size_t)BATCH * LSEQ;            // 4096 rows
  char* p = (char*)d_ws;
  auto alloc = [&](size_t bytes) {
    char* r = p;
    p += (bytes + 255) & ~(size_t)255;
    return r;
  };

  // f32 intermediates
  float* xz    = (float*)alloc(BL * 4096 * 4);
  float* xconv = (float*)alloc(BL * 2048 * 4);
  float* xdbl  = (float*)alloc(BL * 128 * 4);
  float* delta = (float*)alloc(BL * 2048 * 4);
  float* Baug  = (float*)alloc(BL * 32 * 4);
  float* Caug  = (float*)alloc(BL * 32 * 4);
  float* Aaug  = (float*)alloc((size_t)DIN * 32 * 4);
  float* yscan = (float*)alloc(BL * 2048 * 4);
  float* gamma = (float*)alloc(64);
  float* dmean = (float*)alloc(64);
  // bf16 hi/lo split operands
  bf16_t* hid_h = (bf16_t*)alloc(BL * 1024 * 2);
  bf16_t* hid_l = (bf16_t*)alloc(BL * 1024 * 2);
  bf16_t* inw_h = (bf16_t*)alloc((size_t)4096 * 1024 * 2);
  bf16_t* inw_l = (bf16_t*)alloc((size_t)4096 * 1024 * 2);
  bf16_t* xc_h  = (bf16_t*)alloc(BL * 2048 * 2);
  bf16_t* xc_l  = (bf16_t*)alloc(BL * 2048 * 2);
  bf16_t* xpw_h = (bf16_t*)alloc((size_t)128 * 2048 * 2);
  bf16_t* xpw_l = (bf16_t*)alloc((size_t)128 * 2048 * 2);
  bf16_t* xdb_h = (bf16_t*)alloc(BL * 128 * 2);
  bf16_t* xdb_l = (bf16_t*)alloc(BL * 128 * 2);
  bf16_t* dtw_h = (bf16_t*)alloc((size_t)2048 * 64 * 2);
  bf16_t* dtw_l = (bf16_t*)alloc((size_t)2048 * 64 * 2);
  bf16_t* yg_h  = (bf16_t*)alloc(BL * 2048 * 2);
  bf16_t* yg_l  = (bf16_t*)alloc(BL * 2048 * 2);
  bf16_t* ow_h  = (bf16_t*)alloc((size_t)1024 * 2048 * 2);
  bf16_t* ow_l  = (bf16_t*)alloc((size_t)1024 * 2048 * 2);

  prep_kernel<<<1, 64, 0, stream>>>(obs_gain, Dv, gamma, dmean);

  // one-time operand splits (O(elements), hoisted out of GEMM inner loops)
  cvt_kernel<<<16384, 256, 0, stream>>>(hidden, hid_h, hid_l);       // 4096x1024
  cvt_kernel<<<16384, 256, 0, stream>>>(in_proj_w, inw_h, inw_l);    // 4096x1024
  cvt_kernel<<<1024, 256, 0, stream>>>(x_proj_w, xpw_h, xpw_l);      // 128x2048
  cvt_kernel<<<512, 256, 0, stream>>>(dt_proj_w, dtw_h, dtw_l);      // 2048x64
  cvt_kernel<<<8192, 256, 0, stream>>>(out_proj_w, ow_h, ow_l);      // 1024x2048

  // GEMM1: xz(4096x4096) = hidden(4096x1024) x in_proj_w(4096x1024)^T
  {
    int N = 2 * DIN, K = DMOD;
    int tiles = ((int)BL / 32) * (N / 64);           // 8192
    gemm_bf16x3_kernel<<<tiles / 8, 256, 0, stream>>>(
        hid_h, hid_l, DMOD, inw_h, inw_l, DMOD, xz, 2 * DIN,
        nullptr, nullptr, N, K);
  }

  conv_silu_kernel<<<(unsigned)((BL * DIN) / 256), 256, 0, stream>>>(
      xz, conv_w, conv_b, xconv, xc_h, xc_l);

  // GEMM2: xdbl(4096x128) = xconv(4096x2048) x x_proj_w(128x2048)^T
  //        also emits bf16 hi/lo xdbl for GEMM3's A operand
  {
    int N = DTR + 2 * NAUG, K = DIN;
    int tiles = ((int)BL / 32) * (N / 64);           // 256
    gemm_bf16x3_kernel<<<tiles / 8, 256, 0, stream>>>(
        xc_h, xc_l, DIN, xpw_h, xpw_l, DIN, xdbl, N, xdb_h, xdb_l, N, K);
  }

  bcaug_kernel<<<(unsigned)((BL * 32) / 256), 256, 0, stream>>>(
      xdbl, gamma, dmean, Baug, Caug);

  // GEMM3: dt_pre(4096x2048) = dt_low(4096x64, lda=128) x dt_proj_w(2048x64)^T
  {
    int N = DIN, K = DTR;
    int tiles = ((int)BL / 32) * (N / 64);           // 4096
    gemm_bf16x3_kernel<<<tiles / 8, 256, 0, stream>>>(
        xdb_h, xdb_l, DTR + 2 * NAUG, dtw_h, dtw_l, DTR, delta, DIN,
        nullptr, nullptr, N, K);
  }

  delta_kernel<<<(unsigned)((BL * DIN) / 256), 256, 0, stream>>>(delta, dt_proj_b);
  aaug_kernel<<<(unsigned)(((size_t)DIN * 32) / 256), 256, 0, stream>>>(
      A_log, gamma, Aaug);

  scan_kernel<<<(BATCH * DIN) / 8, 256, 0, stream>>>(
      delta, xconv, Baug, Caug, Aaug, yscan);

  gate_kernel<<<(unsigned)((BL * DIN) / 256), 256, 0, stream>>>(
      yscan, xconv, xz, Dv, yg_h, yg_l);

  // GEMM4: out(4096x1024) = ygated(4096x2048) x out_proj_w(1024x2048)^T
  {
    int N = DMOD, K = DIN;
    int tiles = ((int)BL / 32) * (N / 64);           // 2048
    gemm_bf16x3_kernel<<<tiles / 8, 256, 0, stream>>>(
        yg_h, yg_l, DIN, ow_h, ow_l, DIN, (float*)d_out, DMOD,
        nullptr, nullptr, N, K);
  }
}
